// FusedQnA_87282325389584
// MI455X (gfx1250) — matrix operand order repaired and checked by reading, not compiled, not run on hardware
//
#include <hip/hip_runtime.h>

// ---------------- problem constants ----------------
#define IN_F   192
#define HID_F  384
#define HEADS  8
#define HDIM   48
#define KSZ    7
#define PADR   3
#define NB     16
#define HH     56
#define WW     56
#define NPIX   (HH * WW)        // 3136
#define NTILE  (NPIX / 16)      // 196 flat 16-pixel tiles per image

typedef __attribute__((ext_vector_type(16))) __bf16 v16bf;
typedef __attribute__((ext_vector_type(8)))  __bf16 v8bf;
typedef __attribute__((ext_vector_type(8)))  float  v8f;

static __device__ __forceinline__ unsigned short f2bf(float f) {
    unsigned u = __float_as_uint(f);
    u += 0x7FFFu + ((u >> 16) & 1u);          // round-to-nearest-even
    return (unsigned short)(u >> 16);
}
static __device__ __forceinline__ float bf2f(unsigned short s) {
    return __uint_as_float(((unsigned)s) << 16);
}
// Build a v16bf WMMA fragment from two 16-byte-aligned 8-half chunks.
static __device__ __forceinline__ v16bf make16(const unsigned short* p0,
                                               const unsigned short* p1) {
    v8bf lo = *reinterpret_cast<const v8bf*>(p0);
    v8bf hi = *reinterpret_cast<const v8bf*>(p1);
    return __builtin_shufflevector(lo, hi, 0,1,2,3,4,5,6,7,8,9,10,11,12,13,14,15);
}

// ---------------- prep: qWk fold, weight bf16 conversion, exp(rpe) ----------------
__global__ void __launch_bounds__(256)
prep_kernel(const float* __restrict__ query, const float* __restrict__ Wk,
            const float* __restrict__ Wk_bias,
            const float* __restrict__ to_v_w, const float* __restrict__ to_out_w,
            const float* __restrict__ rpe,
            unsigned short* __restrict__ wv_bf, unsigned short* __restrict__ wo_bf,
            float* __restrict__ qWk, float* __restrict__ qb,
            float* __restrict__ rpe_exp) {
    __shared__ float qn[HID_F];
    __shared__ float nrm[HEADS];
    int t = threadIdx.x;
    int gid = blockIdx.x * 256 + t;
    int stride = gridDim.x * 256;
    for (int i = gid; i < HID_F * IN_F;  i += stride) wv_bf[i] = f2bf(to_v_w[i]);
    for (int i = gid; i < HID_F * HID_F; i += stride) wo_bf[i] = f2bf(to_out_w[i]);
    if (blockIdx.x == 0) {
        const float s0 = 1.0f / (6.928203230275509f + 1e-6f);   // 1/(sqrt(48)+eps)
        if (t < HID_F) qn[t] = query[t] * s0;
        __syncthreads();
        if (t < HEADS) {
            float s = 0.f;
            for (int d = 0; d < HDIM; ++d) { float v = qn[t*HDIM + d]; s += v*v; }
            nrm[t] = sqrtf(s) + 1e-6f;
        }
        __syncthreads();
        if (t < HID_F) qn[t] = qn[t] / nrm[t / HDIM];
        __syncthreads();
        for (int i = t; i < IN_F * HEADS; i += 256) {        // qWk[D][h]
            int D = i >> 3, h = i & 7;
            float s = 0.f;
            for (int d = 0; d < HDIM; ++d) s += qn[h*HDIM + d] * Wk[D*HID_F + h*HDIM + d];
            qWk[i] = s;
        }
        if (t < HEADS) {
            float s = 0.f;
            for (int d = 0; d < HDIM; ++d) s += Wk_bias[t*HDIM + d] * qn[t*HDIM + d];
            qb[t] = s;
        }
        for (int i = t; i < HEADS * KSZ * KSZ; i += 256) rpe_exp[i] = __expf(rpe[i]);
    }
}

// ---------------- stage 1: cost_exp + (to_v_w @ x + b) * cost_exp ----------------
#define XT_STRIDE 200   // bf16 elems; row = 400B, 16B aligned

__global__ void __launch_bounds__(256)
stage1_kernel(const float* __restrict__ x, const float* __restrict__ to_v_b,
              const unsigned short* __restrict__ wv_bf,
              const float* __restrict__ qWk, const float* __restrict__ qb,
              float* __restrict__ cost_exp, unsigned short* __restrict__ vexp) {
    __shared__ __align__(16) unsigned short xT[16 * XT_STRIDE];  // [px][D] bf16
    __shared__ float ceT[16 * HEADS];                            // [px][h]
    int t = threadIdx.x;
    int b    = blockIdx.x / NTILE;
    int tile = blockIdx.x % NTILE;
    int p0   = tile * 16;
    const float* xb = x + (size_t)b * IN_F * NPIX + p0;

    for (int i = t; i < IN_F * 16; i += 256) {       // stage x tile, transpose to [px][D]
        int D = i >> 4, j = i & 15;
        xT[j * XT_STRIDE + D] = f2bf(xb[D * NPIX + j]);
    }
    __syncthreads();

    if (t < 128) {                                   // cost projection + exp
        int p = t >> 3, h = t & 7;
        float s = qb[h];
        for (int D = 0; D < IN_F; ++D)
            s += bf2f(xT[p * XT_STRIDE + D]) * qWk[D * HEADS + h];
        float ce = __expf(s);
        ceT[p * HEADS + h] = ce;
        cost_exp[((size_t)b * HEADS + h) * NPIX + p0 + p] = ce;
    }
    __syncthreads();

    int wave = t >> 5, lane = t & 31;
    int n    = lane & 15;
    int lohi = (lane < 16) ? 0 : 1;
    #pragma unroll
    for (int mi = 0; mi < 3; ++mi) {                 // 8 waves * 3 = 24 M-tiles (384 rows)
        int mbase = (wave * 3 + mi) * 16;
        int arow  = mbase + n;
        v8f acc = {};
        #pragma unroll
        for (int kt = 0; kt < IN_F / 32; ++kt) {     // K = 192 -> 6 WMMA steps
            int akb = kt * 32 + lohi * 8;            // A 16x32 bf16 lane layout
            const unsigned short* ap = wv_bf + arow * IN_F + akb;
            v16bf a = make16(ap, ap + 16);
            int bkb = kt * 32 + lohi * 8;            // B 32x16 bf16 lane layout
            const unsigned short* bp = &xT[n * XT_STRIDE + bkb];
            v16bf bv = make16(bp, bp + 16);
            acc = __builtin_amdgcn_wmma_f32_16x16x32_bf16(false, a, false, bv,
                                                          (short)0, acc, false, false);
        }
        int oadd = lohi * 8;
        #pragma unroll
        for (int i = 0; i < 8; ++i) {                // fused bias + *cost_exp, bf16 store
            int o = mbase + i + oadd;
            float val = acc[i] + to_v_b[o];
            val *= ceT[n * HEADS + (o / HDIM)];
            vexp[((size_t)b * HID_F + o) * NPIX + p0 + n] = f2bf(val);
        }
    }
}

// ---------------- stage 2: 7x7 depthwise conv + divide + out GEMM ----------------
#define CCHUNK     32
#define AGG_STRIDE 80    // bf16 elems; row = 160B, 32B aligned

__global__ void __launch_bounds__(256)
stage2_kernel(const float* __restrict__ cost_exp, const unsigned short* __restrict__ vexp,
              const unsigned short* __restrict__ wo_bf, const float* __restrict__ rpe_exp,
              const float* __restrict__ to_out_b, float* __restrict__ out) {
    __shared__ __align__(16) float s_ce[HEADS * 7 * 64];            // cost_exp halo
    __shared__ float s_rcp[HEADS * WW];                             // 1/sum_c
    __shared__ float s_rpe[HEADS * 49];
    __shared__ __align__(16) unsigned short s_chunk[CCHUNK * 7 * 64]; // vexp halo chunk
    __shared__ __align__(32) unsigned short s_aggT[64 * AGG_STRIDE];  // [px][c] bf16 (B tile)
    int t = threadIdx.x;
    int b = blockIdx.x / HH;
    int y = blockIdx.x % HH;

    for (int i = t; i < HEADS * 49; i += 256) s_rpe[i] = rpe_exp[i];
    for (int i = t; i < HEADS * 7 * 64; i += 256) {
        int h = i / 448, rem = i % 448, r = rem >> 6, j = rem & 63;
        int gy = y + r - PADR, gx = j - PADR;
        float v = 0.f;
        if (gy >= 0 && gy < HH && gx >= 0 && gx < WW)
            v = cost_exp[((size_t)b * HEADS + h) * NPIX + gy * WW + gx];
        s_ce[i] = v;
    }
    for (int i = t; i < 64 * AGG_STRIDE; i += 256) s_aggT[i] = 0;   // zero N-pad rows
    __syncthreads();

    for (int i = t; i < HEADS * WW; i += 256) {                     // sum_c -> reciprocal
        int h = i / WW, xx = i % WW;
        float s = 0.f;
        #pragma unroll
        for (int ky = 0; ky < 7; ++ky)
            #pragma unroll
            for (int kx = 0; kx < 7; ++kx)
                s += s_ce[h * 448 + ky * 64 + xx + kx] * s_rpe[h * 49 + ky * 7 + kx];
        s_rcp[i] = 1.0f / s;
    }

    int wave = t >> 5, lane = t & 31;
    int n    = lane & 15;
    int lohi = (lane < 16) ? 0 : 1;
    v8f acc[12] = {};                    // 3 M-tiles x 4 N-tiles per wave, K streamed

    for (int cc = 0; cc < HID_F / CCHUNK; ++cc) {
        __syncthreads();
        for (int i = t; i < CCHUNK * 7 * 64; i += 256) {            // halo chunk of vexp
            int c = i / 448, rem = i % 448, r = rem >> 6, j = rem & 63;
            int gy = y + r - PADR, gx = j - PADR;
            unsigned short v = 0;
            if (gy >= 0 && gy < HH && gx >= 0 && gx < WW)
                v = vexp[((size_t)b * HID_F + cc * CCHUNK + c) * NPIX + gy * WW + gx];
            s_chunk[i] = v;
        }
        __syncthreads();
        for (int i = t; i < CCHUNK * WW; i += 256) {                // dwconv + divide
            int c = i / WW, xx = i % WW;
            int h = (cc * CCHUNK + c) / HDIM;
            float s = 0.f;
            #pragma unroll
            for (int ky = 0; ky < 7; ++ky)
                #pragma unroll
                for (int kx = 0; kx < 7; ++kx)
                    s += bf2f(s_chunk[c * 448 + ky * 64 + xx + kx]) *
                         s_rpe[h * 49 + ky * 7 + kx];
            s_aggT[xx * AGG_STRIDE + c] = f2bf(s * s_rcp[h * WW + xx]);
        }
        __syncthreads();
        int akb = cc * CCHUNK + lohi * 8;
        int bkb = lohi * 8;
        #pragma unroll
        for (int mi = 0; mi < 3; ++mi) {
            int arow = (wave * 3 + mi) * 16 + n;
            const unsigned short* ap = wo_bf + arow * HID_F + akb;
            v16bf a = make16(ap, ap + 16);
            #pragma unroll
            for (int nt = 0; nt < 4; ++nt) {
                const unsigned short* bp = &s_aggT[(nt * 16 + n) * AGG_STRIDE + bkb];
                v16bf bv = make16(bp, bp + 16);
                acc[mi * 4 + nt] = __builtin_amdgcn_wmma_f32_16x16x32_bf16(
                    false, a, false, bv, (short)0, acc[mi * 4 + nt], false, false);
            }
        }
    }

    int oadd = lohi * 8;
    #pragma unroll
    for (int mi = 0; mi < 3; ++mi) {
        int mbase = (wave * 3 + mi) * 16;
        #pragma unroll
        for (int nt = 0; nt < 4; ++nt) {
            int px = nt * 16 + n;
            if (px < WW) {
                #pragma unroll
                for (int i = 0; i < 8; ++i) {
                    int o = mbase + i + oadd;
                    out[((size_t)b * HID_F + o) * NPIX + y * WW + px] =
                        acc[mi * 4 + nt][i] + to_out_b[o];
                }
            }
        }
    }
}

// ---------------- host launcher ----------------
static inline size_t al256(size_t v) { return (v + 255) & ~(size_t)255; }

extern "C" void kernel_launch(void* const* d_in, const int* in_sizes, int n_in,
                              void* d_out, int out_size, void* d_ws, size_t ws_size,
                              hipStream_t stream) {
    const float* x        = (const float*)d_in[0];
    const float* query    = (const float*)d_in[1];
    const float* Wk       = (const float*)d_in[2];
    const float* Wk_bias  = (const float*)d_in[3];
    const float* to_v_w   = (const float*)d_in[4];
    const float* to_v_b   = (const float*)d_in[5];
    const float* to_out_w = (const float*)d_in[6];
    const float* to_out_b = (const float*)d_in[7];
    const float* rpe      = (const float*)d_in[8];
    float* out = (float*)d_out;

    char* ws = (char*)d_ws;
    size_t off = 0;
    unsigned short* wv_bf = (unsigned short*)(ws + off); off = al256(off + (size_t)HID_F*IN_F*2);
    unsigned short* wo_bf = (unsigned short*)(ws + off); off = al256(off + (size_t)HID_F*HID_F*2);
    float* qWk      = (float*)(ws + off); off = al256(off + (size_t)IN_F*HEADS*4);
    float* qb       = (float*)(ws + off); off = al256(off + (size_t)HEADS*4);
    float* rpe_exp  = (float*)(ws + off); off = al256(off + (size_t)HEADS*49*4);
    float* cost_exp = (float*)(ws + off); off = al256(off + (size_t)NB*HEADS*NPIX*4);
    unsigned short* vexp = (unsigned short*)(ws + off);
    off = al256(off + (size_t)NB*HID_F*NPIX*2);
    (void)ws_size; (void)n_in; (void)in_sizes; (void)out_size;

    prep_kernel<<<128, 256, 0, stream>>>(query, Wk, Wk_bias, to_v_w, to_out_w, rpe,
                                         wv_bf, wo_bf, qWk, qb, rpe_exp);
    stage1_kernel<<<NB * NTILE, 256, 0, stream>>>(x, to_v_b, wv_bf, qWk, qb,
                                                  cost_exp, vexp);
    stage2_kernel<<<NB * HH, 256, 0, stream>>>(cost_exp, vexp, wo_bf, rpe_exp,
                                               to_out_b, out);
}
